// ParagraphGATInference_59485297050282
// MI455X (gfx1250) — matrix operand order repaired and checked by reading, not compiled, not run on hardware
//
#include <hip/hip_runtime.h>
#include <hip/hip_bf16.h>

typedef __attribute__((ext_vector_type(16))) _Float16 v16h;
typedef __attribute__((ext_vector_type(8)))  _Float16 v8h;
typedef __attribute__((ext_vector_type(4)))  _Float16 v4h;
typedef __attribute__((ext_vector_type(8)))  float    v8f;

#define DIM 128

// ======================= setup: CSR by destination =======================
__global__ void k_init_counts(int* cnt, int n) {
  int t = blockIdx.x * blockDim.x + threadIdx.x;
  if (t < n) cnt[t] = 1;  // self loop contributes 1 in-edge per node
}
__global__ void k_count_edges(const int* __restrict__ dst, int* cnt, int e) {
  int t = blockIdx.x * blockDim.x + threadIdx.x;
  if (t < e) atomicAdd(&cnt[dst[t]], 1);
}
__global__ void k_scan(const int* __restrict__ cnt, int* rowptr, int n) {
  __shared__ int sm[1024];
  __shared__ int carry;
  int t = threadIdx.x;
  if (t == 0) carry = 0;
  __syncthreads();
  for (int base = 0; base < n; base += 1024) {
    int v = (base + t < n) ? cnt[base + t] : 0;
    sm[t] = v;
    __syncthreads();
    for (int off = 1; off < 1024; off <<= 1) {
      int x = (t >= off) ? sm[t - off] : 0;
      __syncthreads();
      sm[t] += x;
      __syncthreads();
    }
    if (base + t < n) rowptr[base + t] = carry + sm[t] - v;  // exclusive
    __syncthreads();
    if (t == 0) carry += sm[1023];
    __syncthreads();
  }
  if (t == 0) rowptr[n] = carry;
}
__global__ void k_fill_self(const int* __restrict__ rowptr, int* cursor,
                            int* csr_src, int* csr_dst, int n) {
  int t = blockIdx.x * blockDim.x + threadIdx.x;
  if (t < n) {
    int p = rowptr[t];
    csr_src[p] = t; csr_dst[p] = t;
    cursor[t] = p + 1;
  }
}
__global__ void k_fill_edges(const int* __restrict__ src, const int* __restrict__ dst,
                             int* cursor, int* csr_src, int* csr_dst, int e) {
  int t = blockIdx.x * blockDim.x + threadIdx.x;
  if (t < e) {
    int d = dst[t];
    int p = atomicAdd(&cursor[d], 1);
    csr_src[p] = src[t]; csr_dst[p] = d;
  }
}

// ======================= conversions =======================
__global__ void k_cvt_f16(const float* __restrict__ in, _Float16* __restrict__ out, int n) {
  int t = blockIdx.x * blockDim.x + threadIdx.x;
  if (t < n) out[t] = (_Float16)in[t];
}
// W: [128, cols] f32 row-major  ->  WT: [cols, 128] f16 row-major
__global__ void k_transpose_cvt(const float* __restrict__ W, _Float16* __restrict__ WT, int cols) {
  int t = blockIdx.x * blockDim.x + threadIdx.x;
  if (t < DIM * cols) {
    int r = t / cols, c = t - r * cols;
    WT[(size_t)c * DIM + r] = (_Float16)W[t];
  }
}

// ======================= WMMA GEMM: C[M,Nout] = A[M,128] * B[128,Nout] =======================
// A row-major f16, B supplied transposed (BT[Nout,128] f16). One wave computes a 32x64 tile
// (2 A-fragments x 4 B-fragments = 8 accumulators, 32 WMMAs/wave): every B fragment feeds two
// independent WMMAs, halving exposed load latency per matrix op vs a 16x64 tile.
// Fragment layouts per CDNA5 ISA 7.12.2:
//   A 16x32 f16 : lane L (mr=L&15, hi=L>>4) holds row mr, elems e=0..7 -> K=k0+8*hi+e,
//                 e=8..15 -> K=k0+16+8*hi+(e-8)   (two contiguous 16B chunks)
//   B 32x16 f16 : lane L holds col (L&15), same K mapping as A (two 16B chunks from BT row)
//   C/D 16x16 f32: VGPR r -> row r+8*hi, col L&15
__global__ void k_gemm_wmma(const _Float16* __restrict__ A,
                            const _Float16* __restrict__ BT,
                            _Float16* __restrict__ C,
                            int M, int Nout) {
  int wave = blockIdx.x * (blockDim.x >> 5) + (threadIdx.x >> 5);
  int lane = threadIdx.x & 31;
  int tilesM = (M + 31) >> 5;   // 32 rows per wave
  int tilesN = Nout >> 6;       // 64 cols per wave
  if (wave >= tilesM * tilesN) return;
  int tm = wave % tilesM;
  int tn = wave / tilesM;
  int m0 = tm << 5, n0 = tn << 6;
  int mr = lane & 15, hi = lane >> 4;

  int rowA0 = m0 + mr;      if (rowA0 >= M) rowA0 = M - 1;
  int rowA1 = m0 + 16 + mr; if (rowA1 >= M) rowA1 = M - 1;
  const _Float16* arow0 = A + (size_t)rowA0 * DIM;
  const _Float16* arow1 = A + (size_t)rowA1 * DIM;

  v8f acc[2][4] = {};
#pragma unroll
  for (int k0 = 0; k0 < DIM; k0 += 32) {
    union { v16h v; v8h h[2]; } a0, a1;
    a0.h[0] = *(const v8h*)(arow0 + k0 + 8 * hi);
    a0.h[1] = *(const v8h*)(arow0 + k0 + 16 + 8 * hi);
    a1.h[0] = *(const v8h*)(arow1 + k0 + 8 * hi);
    a1.h[1] = *(const v8h*)(arow1 + k0 + 16 + 8 * hi);
#pragma unroll
    for (int t = 0; t < 4; ++t) {
      const _Float16* bcol = BT + (size_t)(n0 + t * 16 + mr) * DIM + k0 + 8 * hi;
      union { v16h v; v8h h[2]; } b;
      b.h[0] = *(const v8h*)(bcol);
      b.h[1] = *(const v8h*)(bcol + 16);
      acc[0][t] = __builtin_amdgcn_wmma_f32_16x16x32_f16(
          false, a0.v, false, b.v, (short)0, acc[0][t], false, false);
      acc[1][t] = __builtin_amdgcn_wmma_f32_16x16x32_f16(
          false, a1.v, false, b.v, (short)0, acc[1][t], false, false);
    }
  }
#pragma unroll
  for (int am = 0; am < 2; ++am) {
#pragma unroll
    for (int t = 0; t < 4; ++t) {
      int col = n0 + t * 16 + mr;
#pragma unroll
      for (int r = 0; r < 8; ++r) {
        int row = m0 + am * 16 + r + 8 * hi;
        if (row < M) C[(size_t)row * Nout + col] = (_Float16)acc[am][t][r];
      }
    }
  }
}

// ======================= edge attention scores =======================
// one wave per CSR edge slot; lane owns 4 channels; wave32 shuffle reduction per head
template<int H>
__global__ void k_edge_scores(const _Float16* __restrict__ xl, const _Float16* __restrict__ xr,
                              const int* __restrict__ csr_src, const int* __restrict__ csr_dst,
                              const float* __restrict__ att, float* __restrict__ e_buf, int etot) {
  int wave = blockIdx.x * (blockDim.x >> 5) + (threadIdx.x >> 5);
  int lane = threadIdx.x & 31;
  if (wave >= etot) return;
  int s = csr_src[wave], d = csr_dst[wave];
  const _Float16* pl = xl + (size_t)s * H * DIM + lane * 4;
  const _Float16* pr = xr + (size_t)d * H * DIM + lane * 4;
#pragma unroll
  for (int h = 0; h < H; ++h) {
    v4h l = *(const v4h*)(pl + h * DIM);
    v4h r = *(const v4h*)(pr + h * DIM);
    float dot = 0.f;
#pragma unroll
    for (int q = 0; q < 4; ++q) {
      float z = (float)l[q] + (float)r[q];
      z = z > 0.f ? z : 0.2f * z;                 // leaky_relu slope 0.2
      dot += z * att[h * DIM + lane * 4 + q];
    }
#pragma unroll
    for (int off = 16; off > 0; off >>= 1) dot += __shfl_xor(dot, off, 32);
    if (lane == 0) e_buf[(size_t)wave * H + h] = dot;
  }
}

// ======================= per-(dst,head) segment softmax (in-place -> alpha) ==============
__global__ void k_softmax(float* e_buf, const int* __restrict__ rowptr, int n, int H) {
  int tid = blockIdx.x * blockDim.x + threadIdx.x;
  if (tid >= n * H) return;
  int i = tid / H, h = tid - i * H;
  int j0 = rowptr[i], j1 = rowptr[i + 1];
  float m = -1e30f;
  for (int j = j0; j < j1; ++j) { float v = e_buf[(size_t)j * H + h]; m = v > m ? v : m; }
  float s = 0.f;
  for (int j = j0; j < j1; ++j) s += __expf(e_buf[(size_t)j * H + h] - m);
  float inv = 1.f / s;
  for (int j = j0; j < j1; ++j) e_buf[(size_t)j * H + h] = __expf(e_buf[(size_t)j * H + h] - m) * inv;
}

// ======================= aggregation + head-mean + bias (+relu) + residual ================
// one wave per destination node; lane owns 4 channels across all H heads (H*4 f32 regs)
template<int H>
__global__ void k_aggregate(const _Float16* __restrict__ xl, const float* __restrict__ alpha,
                            const int* __restrict__ rowptr, const int* __restrict__ csr_src,
                            const float* __restrict__ bias, const float* __restrict__ hprev,
                            float* __restrict__ hnext, int n, int do_relu) {
  int wave = blockIdx.x * (blockDim.x >> 5) + (threadIdx.x >> 5);
  int lane = threadIdx.x & 31;
  if (wave >= n) return;
  int j0 = rowptr[wave], j1 = rowptr[wave + 1];
  float acc[H][4];
#pragma unroll
  for (int h = 0; h < H; ++h) { acc[h][0] = acc[h][1] = acc[h][2] = acc[h][3] = 0.f; }
  for (int j = j0; j < j1; ++j) {
    int s = csr_src[j];
    const _Float16* pl = xl + (size_t)s * H * DIM + lane * 4;
#pragma unroll
    for (int h = 0; h < H; ++h) {
      float a = alpha[(size_t)j * H + h];
      v4h v = *(const v4h*)(pl + h * DIM);
      acc[h][0] += a * (float)v[0];
      acc[h][1] += a * (float)v[1];
      acc[h][2] += a * (float)v[2];
      acc[h][3] += a * (float)v[3];
    }
  }
  const float invH = 1.f / (float)H;
#pragma unroll
  for (int q = 0; q < 4; ++q) {
    float m = 0.f;
#pragma unroll
    for (int h = 0; h < H; ++h) m += acc[h][q];
    m *= invH;
    int c = lane * 4 + q;
    float val = m + bias[c];
    if (do_relu) val = fmaxf(val, 0.f);
    val += hprev[(size_t)wave * DIM + c];
    hnext[(size_t)wave * DIM + c] = val;
  }
}

// ======================= host launcher =======================
extern "C" void kernel_launch(void* const* d_in, const int* in_sizes, int n_in,
                              void* d_out, int out_size, void* d_ws, size_t ws_size,
                              hipStream_t stream) {
  const float* x = (const float*)d_in[0];
  const int* ei = (const int*)d_in[1];
  const float* Wl[3]  = {(const float*)d_in[2], (const float*)d_in[6],  (const float*)d_in[10]};
  const float* Wr[3]  = {(const float*)d_in[3], (const float*)d_in[7],  (const float*)d_in[11]};
  const float* att[3] = {(const float*)d_in[4], (const float*)d_in[8],  (const float*)d_in[12]};
  const float* bia[3] = {(const float*)d_in[5], (const float*)d_in[9],  (const float*)d_in[13]};

  int N = in_sizes[0] / DIM;
  int E = in_sizes[1] / 2;
  int Et = E + N;                    // self loops appended
  int H[3] = {in_sizes[4] / DIM, in_sizes[8] / DIM, in_sizes[12] / DIM};
  const int* src = ei;
  const int* dst = ei + E;

  // ---- carve workspace ----
  char* base = (char*)d_ws;
  size_t off = 0;
  auto carve = [&](size_t bytes) -> char* {
    char* p = base + off;
    off = (off + bytes + 255) & ~(size_t)255;
    return p;
  };
  int*      cnt     = (int*)carve((size_t)N * 4);
  int*      rowptr  = (int*)carve((size_t)(N + 1) * 4);
  int*      cursor  = (int*)carve((size_t)N * 4);
  int*      csr_src = (int*)carve((size_t)Et * 4);
  int*      csr_dst = (int*)carve((size_t)Et * 4);
  float*    e_buf   = (float*)carve((size_t)Et * 8 * 4);
  _Float16* xl      = (_Float16*)carve((size_t)N * 8 * DIM * 2);
  _Float16* xr      = (_Float16*)carve((size_t)N * 8 * DIM * 2);
  float*    hA      = (float*)carve((size_t)N * DIM * 4);
  float*    hB      = (float*)carve((size_t)N * DIM * 4);
  _Float16* hf16    = (_Float16*)carve((size_t)N * DIM * 2);
  _Float16* WTl     = (_Float16*)carve((size_t)8 * DIM * DIM * 2);
  _Float16* WTr     = (_Float16*)carve((size_t)8 * DIM * DIM * 2);
  if (off > ws_size) return;  // workspace too small to proceed safely

  const int TB = 256;
  // ---- build CSR (once; graph identical across layers) ----
  k_init_counts<<<(N + TB - 1) / TB, TB, 0, stream>>>(cnt, N);
  k_count_edges<<<(E + TB - 1) / TB, TB, 0, stream>>>(dst, cnt, E);
  k_scan<<<1, 1024, 0, stream>>>(cnt, rowptr, N);
  k_fill_self<<<(N + TB - 1) / TB, TB, 0, stream>>>(rowptr, cursor, csr_src, csr_dst, N);
  k_fill_edges<<<(E + TB - 1) / TB, TB, 0, stream>>>(src, dst, cursor, csr_src, csr_dst, E);

  const float* xin[3] = {x, hA, hB};
  float* xout[3] = {hA, hB, (float*)d_out};
  int relu[3] = {1, 1, 0};

  for (int li = 0; li < 3; ++li) {
    int h = H[li];
    int Kout = h * DIM;
    k_cvt_f16<<<(N * DIM + TB - 1) / TB, TB, 0, stream>>>(xin[li], hf16, N * DIM);
    k_transpose_cvt<<<(DIM * Kout + TB - 1) / TB, TB, 0, stream>>>(Wl[li], WTl, Kout);
    k_transpose_cvt<<<(DIM * Kout + TB - 1) / TB, TB, 0, stream>>>(Wr[li], WTr, Kout);

    int tilesM = (N + 31) / 32, tilesN = Kout / 64;
    int gblocks = (tilesM * tilesN + 7) / 8;
    k_gemm_wmma<<<gblocks, 256, 0, stream>>>(hf16, WTl, xl, N, Kout);
    k_gemm_wmma<<<gblocks, 256, 0, stream>>>(hf16, WTr, xr, N, Kout);

    int eblocks = (Et + 7) / 8;
    int nblocks = (N + 7) / 8;
    int sblocks = (N * h + TB - 1) / TB;
    if (h == 4) {
      k_edge_scores<4><<<eblocks, 256, 0, stream>>>(xl, xr, csr_src, csr_dst, att[li], e_buf, Et);
      k_softmax<<<sblocks, TB, 0, stream>>>(e_buf, rowptr, N, h);
      k_aggregate<4><<<nblocks, 256, 0, stream>>>(xl, e_buf, rowptr, csr_src, bia[li],
                                                  xin[li], xout[li], N, relu[li]);
    } else {
      k_edge_scores<8><<<eblocks, 256, 0, stream>>>(xl, xr, csr_src, csr_dst, att[li], e_buf, Et);
      k_softmax<<<sblocks, TB, 0, stream>>>(e_buf, rowptr, N, h);
      k_aggregate<8><<<nblocks, 256, 0, stream>>>(xl, e_buf, rowptr, csr_src, bia[li],
                                                  xin[li], xout[li], N, relu[li]);
    }
  }
}
